// Conv1d_52853867545068
// MI455X (gfx1250) — matrix operand order repaired and checked by reading, not compiled, not run on hardware
//
#include <hip/hip_runtime.h>
#include <math.h>

// Problem constants (from reference)
constexpr int NB = 8;      // batch
constexpr int NC = 256;    // in channels
constexpr int NO = 256;    // out channels
constexpr int NT = 8192;   // time
constexpr int NK = 9;      // kernel width
constexpr int PAD = 4;
constexpr int TPAD = NT + 16;  // padded time (data at rows [8, NT+8))
constexpr int LWS = 272;       // LDS row stride in halves (544B, breaks bank conflicts)

typedef _Float16 v16h __attribute__((ext_vector_type(16)));
typedef _Float16 v8h  __attribute__((ext_vector_type(8)));
typedef float    v8f  __attribute__((ext_vector_type(8)));
typedef int      v4i  __attribute__((ext_vector_type(4)));

// Address-space qualified pointer types for the async global->LDS builtin:
// expected param type (from compiler diagnostic) is int4 in device/global AS.
typedef __attribute__((address_space(1))) v4i gv4i;
typedef __attribute__((address_space(3))) v4i lv4i;

#if defined(__has_builtin)
#if __has_builtin(__builtin_amdgcn_global_load_async_to_lds_b128)
#define HAVE_ASYNC_LDS 1
#endif
#if __has_builtin(__builtin_amdgcn_s_wait_asynccnt)
#define HAVE_WAIT_ASYNC 1
#endif
#endif

__device__ __forceinline__ void async_copy_b128(const _Float16* gsrc, _Float16* lds_dst) {
#if defined(HAVE_ASYNC_LDS)
    __builtin_amdgcn_global_load_async_to_lds_b128(
        (gv4i*)(void*)gsrc, (lv4i*)(void*)lds_dst, 0, 0);
#else
    *(v8h*)lds_dst = *(const v8h*)gsrc;   // fallback: load + ds_store
#endif
}

__device__ __forceinline__ void wait_async_lds() {
#if defined(HAVE_ASYNC_LDS)
#if defined(HAVE_WAIT_ASYNC)
    __builtin_amdgcn_s_wait_asynccnt(0);
#else
    asm volatile("s_wait_asynccnt 0x0" ::: "memory");
#endif
#endif
}

// ---------------------------------------------------------------------------
// Kernel 1: scale weights by sqrt(|alpha_k|)/sqrt(C), transpose (O,C,K)->(K,O,C),
// cast to f16. Three weight tensors handled together.
// ---------------------------------------------------------------------------
__global__ void prep_weights(const float* __restrict__ weight,
                             const float* __restrict__ w0,
                             const float* __restrict__ w,
                             const float* __restrict__ alpha,
                             _Float16* __restrict__ wv_t,
                             _Float16* __restrict__ w0_t,
                             _Float16* __restrict__ ww_t) {
    int idx = blockIdx.x * blockDim.x + threadIdx.x;
    if (idx >= NO * NC * NK) return;
    int o   = idx / (NC * NK);
    int rem = idx - o * (NC * NK);
    int c   = rem / NK;
    int k   = rem - c * NK;
    float s = sqrtf(fabsf(alpha[k])) * 0.0625f;  // 1/sqrt(256)
    int dst = (k * NO + o) * NC + c;
    wv_t[dst] = (_Float16)(weight[idx] * s);
    w0_t[dst] = (_Float16)(w0[idx] * s);
    ww_t[dst] = (_Float16)(w[idx] * s);
}

// ---------------------------------------------------------------------------
// Kernel 2: de-interleave 3 tracks, relu / heaviside mask, cast to f16,
// store time-padded (B, TPAD, C) with zero halo.
// ---------------------------------------------------------------------------
__global__ void prep_acts(const float* __restrict__ x,
                          _Float16* __restrict__ axv,
                          _Float16* __restrict__ ax0,
                          _Float16* __restrict__ adx) {
    long idx = (long)blockIdx.x * blockDim.x + threadIdx.x;  // over B*TPAD*C
    if (idx >= (long)NB * TPAD * NC) return;
    int  c   = (int)(idx % NC);
    long rem = idx / NC;
    int  tp  = (int)(rem % TPAD);
    int  b   = (int)(rem / TPAD);
    float xv = 0.f, x0 = 0.f, dx = 0.f;
    if (tp >= 8 && tp < NT + 8) {
        int t = tp - 8;
        long src = (((long)b * NC + c) * NT + t) * 3;
        float v0 = x[src + 0];
        float v1 = x[src + 1];
        float v2 = x[src + 2];
        xv = fmaxf(v0, 0.f);
        x0 = fmaxf(v1, 0.f);
        dx = (v1 >= 0.f) ? v2 : 0.f;
    }
    axv[idx] = (_Float16)xv;
    ax0[idx] = (_Float16)x0;
    adx[idx] = (_Float16)dx;
}

// ---------------------------------------------------------------------------
// Kernel 3: implicit-GEMM conv via v_wmma_f32_16x16x32_f16.
// Block = 4 waves; each wave owns a 16(o) x 32(t) macro-tile (two 16x16 tiles)
// of all three output tracks. Weights for the current k are staged block-wide
// into LDS via async global->LDS, A fragments reused across both t-tiles.
// ---------------------------------------------------------------------------
struct F3 { float x, y, z; };

__global__ __launch_bounds__(128)
void conv_wmma(const _Float16* __restrict__ axv,
               const _Float16* __restrict__ ax0,
               const _Float16* __restrict__ adx,
               const _Float16* __restrict__ wv_t,
               const _Float16* __restrict__ w0_t,
               const _Float16* __restrict__ ww_t,
               const float* __restrict__ bias,
               const float* __restrict__ b0v,
               const float* __restrict__ bv,
               const float* __restrict__ beta,
               float* __restrict__ out) {
    __shared__ _Float16 lw[3 * 16 * LWS];   // 3 weight tensors, 16 o-rows, padded

    const int tid   = threadIdx.x & 127;     // provably < 128: no guard code below
    const int lane  = tid & 31;
    const int wave  = tid >> 5;
    const int tbase = blockIdx.x * 128 + wave * 32;  // this wave: t in [tbase, tbase+32)
    const int o0    = blockIdx.y * 16;
    const int b     = blockIdx.z;

    const int half = lane >> 4;   // 0: lanes 0-15, 1: lanes 16-31
    const int lid  = lane & 15;

    // k order used by this code (contiguous k = 16*half + s, same for A and B):
    // A 16x32 f16: lanes 0-15 hold row M, K 0-15; lanes 16-31 K 16-31
    const int aoff = half * 16;
    // B 32x16 f16: lanes 0-15 hold col N, K 0-15; lanes 16-31 K 16-31
    const int boff = half * 16;

    v8f accx0 = {}, acc00 = {}, accd0 = {};   // t-tile 0
    v8f accx1 = {}, acc01 = {}, accd1 = {};   // t-tile 1

    const long actB = (long)b * TPAD * NC;
    const _Float16* __restrict__ axvB = axv + actB;
    const _Float16* __restrict__ ax0B = ax0 + actB;
    const _Float16* __restrict__ adxB = adx + actB;

    for (int k = 0; k < NK; ++k) {
        // ---- cooperative stage of this k's weight tile (3 x 16 x 256 halves) ----
        if (k) __syncthreads();               // previous k's LDS reads complete
        {
            const long koff = ((long)k * NO + o0) * NC;
            #pragma unroll
            for (int j = 0; j < 3; ++j) {
                const _Float16* src = (j == 0) ? (wv_t + koff)
                                    : (j == 1) ? (w0_t + koff)
                                               : (ww_t + koff);
                #pragma unroll
                for (int it = 0; it < 4; ++it) {           // 512 granules / 128 thr
                    const int g  = tid + it * 128;         // < 512 provably
                    const int oi = g >> 5;                 // row (o)
                    const int ci = (g & 31) << 3;          // col (c), in halves
                    async_copy_b128(src + oi * NC + ci,
                                    &lw[(j * 16 + oi) * LWS + ci]);
                }
            }
        }
        wait_async_lds();
        __syncthreads();                      // staged tile visible to all waves

        const long brow0 = (long)(tbase + lid + k + PAD) * NC;  // t-tile 0 act row
        const long brow1 = brow0 + 16 * (long)NC;               // t-tile 1 act row

        // speculative prefetch of next k's activation rows
        if (k + 1 < NK) {
            __builtin_prefetch(axvB + brow0 + NC, 0, 1);
            __builtin_prefetch(ax0B + brow0 + NC, 0, 1);
            __builtin_prefetch(adxB + brow0 + NC, 0, 1);
        }

        const int arow = lid * LWS;
        #pragma unroll
        for (int c0 = 0; c0 < NC; c0 += 32) {
            // ---- A fragments from LDS (ds_load_b128 x6), shared across t-tiles ----
            const int wb = arow + c0 + aoff;
            v8h wvl = *(const v8h*)&lw[wb];
            v8h wvh = *(const v8h*)&lw[wb + 8];
            v8h w0l = *(const v8h*)&lw[16 * LWS + wb];
            v8h w0h = *(const v8h*)&lw[16 * LWS + wb + 8];
            v8h wwl = *(const v8h*)&lw[32 * LWS + wb];
            v8h wwh = *(const v8h*)&lw[32 * LWS + wb + 8];
            v16h Awv = __builtin_shufflevector(wvl, wvh, 0,1,2,3,4,5,6,7,8,9,10,11,12,13,14,15);
            v16h Aw0 = __builtin_shufflevector(w0l, w0h, 0,1,2,3,4,5,6,7,8,9,10,11,12,13,14,15);
            v16h Aww = __builtin_shufflevector(wwl, wwh, 0,1,2,3,4,5,6,7,8,9,10,11,12,13,14,15);

            // ---- B fragments from global (32B each), two t-tiles x 3 tracks ----
            const long ab0 = brow0 + c0 + boff;
            const long ab1 = brow1 + c0 + boff;
            v16h Bxv0 = *(const v16h*)(axvB + ab0);
            v16h Bx00 = *(const v16h*)(ax0B + ab0);
            v16h Bdx0 = *(const v16h*)(adxB + ab0);
            v16h Bxv1 = *(const v16h*)(axvB + ab1);
            v16h Bx01 = *(const v16h*)(ax0B + ab1);
            v16h Bdx1 = *(const v16h*)(adxB + ab1);

            // ---- 8 WMMAs: x += Wv*Xv ; x0 += W0*X0 ; dx += W0*Xd + Ww*X0 ----
            accx0 = __builtin_amdgcn_wmma_f32_16x16x32_f16(false, Awv, false, Bxv0,
                                                           (short)0, accx0, false, false);
            accx1 = __builtin_amdgcn_wmma_f32_16x16x32_f16(false, Awv, false, Bxv1,
                                                           (short)0, accx1, false, false);
            acc00 = __builtin_amdgcn_wmma_f32_16x16x32_f16(false, Aw0, false, Bx00,
                                                           (short)0, acc00, false, false);
            acc01 = __builtin_amdgcn_wmma_f32_16x16x32_f16(false, Aw0, false, Bx01,
                                                           (short)0, acc01, false, false);
            accd0 = __builtin_amdgcn_wmma_f32_16x16x32_f16(false, Aw0, false, Bdx0,
                                                           (short)0, accd0, false, false);
            accd1 = __builtin_amdgcn_wmma_f32_16x16x32_f16(false, Aw0, false, Bdx1,
                                                           (short)0, accd1, false, false);
            accd0 = __builtin_amdgcn_wmma_f32_16x16x32_f16(false, Aww, false, Bx00,
                                                           (short)0, accd0, false, false);
            accd1 = __builtin_amdgcn_wmma_f32_16x16x32_f16(false, Aww, false, Bx01,
                                                           (short)0, accd1, false, false);
        }
    }

    // Epilogue: + bias*sqrt(|beta|); packed 12B stores into (B,O,T,3)
    const float sb = sqrtf(fabsf(beta[0]));
    const int t0 = tbase + lid;
    #pragma unroll
    for (int r = 0; r < 8; ++r) {
        // D layout: VGPR r -> M = r + half*8 ; N = lane%16
        const int o  = o0 + r + half * 8;
        const float bx = bias[o] * sb;
        const float b0 = b0v[o]  * sb;
        const float bd = bv[o]   * sb;
        const long base = ((long)b * NO + o) * NT;
        F3 v0 = { accx0[r] + bx, acc00[r] + b0, accd0[r] + bd };
        F3 v1 = { accx1[r] + bx, acc01[r] + b0, accd1[r] + bd };
        *(F3*)(out + (base + t0) * 3)      = v0;
        *(F3*)(out + (base + t0 + 16) * 3) = v1;
    }
}

// ---------------------------------------------------------------------------
extern "C" void kernel_launch(void* const* d_in, const int* in_sizes, int n_in,
                              void* d_out, int out_size, void* d_ws, size_t ws_size,
                              hipStream_t stream) {
    const float* x      = (const float*)d_in[0];  // (B,C,T,3)
    const float* weight = (const float*)d_in[1];  // (O,C,K)
    const float* w0     = (const float*)d_in[2];  // (O,C,K)
    const float* w      = (const float*)d_in[3];  // (O,C,K)
    const float* alpha  = (const float*)d_in[4];  // (1,1,K)
    const float* bias   = (const float*)d_in[5];  // (O)
    const float* b0v    = (const float*)d_in[6];  // (O)
    const float* bv     = (const float*)d_in[7];  // (O)
    const float* beta   = (const float*)d_in[8];  // scalar
    float* out = (float*)d_out;                   // (B,O,T,3)

    // Workspace layout (f16): 3 padded activation tensors + 3 transposed weights
    char* ws = (char*)d_ws;
    const size_t actElems = (size_t)NB * TPAD * NC;
    const size_t wElems   = (size_t)NK * NO * NC;
    _Float16* axv  = (_Float16*)ws;  ws += actElems * sizeof(_Float16);
    _Float16* ax0  = (_Float16*)ws;  ws += actElems * sizeof(_Float16);
    _Float16* adx  = (_Float16*)ws;  ws += actElems * sizeof(_Float16);
    _Float16* wv_t = (_Float16*)ws;  ws += wElems * sizeof(_Float16);
    _Float16* w0_t = (_Float16*)ws;  ws += wElems * sizeof(_Float16);
    _Float16* ww_t = (_Float16*)ws;

    {
        int n = NO * NC * NK;
        prep_weights<<<(n + 255) / 256, 256, 0, stream>>>(weight, w0, w, alpha,
                                                          wv_t, w0_t, ww_t);
    }
    {
        long n = (long)actElems;
        prep_acts<<<(unsigned)((n + 255) / 256), 256, 0, stream>>>(x, axv, ax0, adx);
    }
    {
        dim3 grid(NT / 128, NO / 16, NB);   // 4 waves/block, 32 t each
        conv_wmma<<<grid, 128, 0, stream>>>(axv, ax0, adx, wv_t, w0_t, ww_t,
                                            bias, b0v, bv, beta, out);
    }
}
